// VectorQuantizer_57466662420709
// MI455X (gfx1250) — matrix operand order repaired and checked by reading, not compiled, not run on hardware
//
#include <hip/hip_runtime.h>
#include <hip/hip_bf16.h>

// ---------------------------------------------------------------------------
// VectorQuantizer for MI455X (gfx1250, wave32, WMMA)
//   x:        [16, 256, 4096] f32
//   codebook: [1024, 256]     f32
//   out:      quant [16,256,4096] f32  ++  loss scalar f32
//
// Pipeline: pre-scrambled f16 codebook in WMMA B-layout, double-buffered
// async global->LDS staging overlapped with v_wmma_f32_16x16x32_f16 sweep,
// shuffle argmin, coalesced gather/write with deterministic loss reduction.
// ---------------------------------------------------------------------------

typedef __attribute__((ext_vector_type(16))) _Float16 v16h;
typedef __attribute__((ext_vector_type(8)))  float    v8f;

// async-to-LDS builtin operand types
typedef __attribute__((__vector_size__(16))) int vs4i;
typedef __attribute__((address_space(1))) vs4i gvs4i;   // global int4
typedef __attribute__((address_space(3))) vs4i lvs4i;   // LDS int4

#define B_DIM 16
#define C_DIM 256
#define T_DIM 4096
#define K_CODES 1024
#define N_ROWS (B_DIM * T_DIM)          // 65536
#define ROWS_PER_WG 128                 // 8 waves * 16 rows
#define CHUNK_CODES 64                  // codes staged in LDS per iteration
#define CHUNK_HALFS (CHUNK_CODES * C_DIM)   // 16384 halfs = 32KB
#define N_CHUNKS (K_CODES / CHUNK_CODES)    // 16

#if defined(__gfx1250__) && __has_builtin(__builtin_amdgcn_global_load_async_to_lds_b128)
#define HAVE_ASYNC_LDS 1
#endif

// ------------------------- prep: scramble codebook -------------------------
// Pre-pack codebook as f16 in the B-operand order this kernel uses (contiguous k, same as A):
//   B (32x16, 16-bit): lanes 0-15 hold K=0..15, lanes 16-31 hold K=16..31,
//   VGPR v packs (K=2v, 2v+1)  =>  element e of v16h maps to K = 16*hi + e.
// Scrambled index: (((ct*8 + kc)*32 + lane)*16 + e), ct = code tile (16 codes),
// kc = k-chunk of 32.
__global__ __launch_bounds__(256) void vq_prep_scramble(
    const float* __restrict__ cb, _Float16* __restrict__ cbh) {
  int s    = blockIdx.x * 256 + threadIdx.x;   // 0 .. 262143
  int e    = s & 15;
  int lane = (s >> 4) & 31;
  int kc   = (s >> 9) & 7;
  int ct   = s >> 12;
  int n    = ct * 16 + (lane & 15);
  int hi   = lane >> 4;
  int c    = kc * 32 + hi * 16 + e;
  cbh[s] = (_Float16)cb[n * C_DIM + c];
}

// ------------------------- prep: |e_k|^2 ------------------------------------
__global__ __launch_bounds__(64) void vq_prep_qnorm(
    const float* __restrict__ cb, float* __restrict__ qn) {
  int code = blockIdx.x;
  const float4* p = (const float4*)(cb + (size_t)code * C_DIM);
  float4 f = p[threadIdx.x];
  float ss = f.x * f.x + f.y * f.y + f.z * f.z + f.w * f.w;
#pragma unroll
  for (int off = 16; off; off >>= 1) ss += __shfl_xor(ss, off, 32);
  __shared__ float w[2];
  if ((threadIdx.x & 31) == 0) w[threadIdx.x >> 5] = ss;
  __syncthreads();
  if (threadIdx.x == 0) qn[code] = w[0] + w[1];
}

// ------------------------- main kernel --------------------------------------
__global__ __launch_bounds__(256) void vq_main(
    const float*    __restrict__ x,
    const float*    __restrict__ cb,    // original f32 codebook (for gather)
    const _Float16* __restrict__ cbh,   // scrambled f16 codebook
    const float*    __restrict__ qn,    // |e_k|^2
    float*          __restrict__ out,   // quant [B,C,T]
    float*          __restrict__ part)  // per-block loss partials
{
  __shared__ _Float16 lds_cb[2][CHUNK_HALFS] __attribute__((aligned(32))); // 64KB
  __shared__ float lds_q[K_CODES];                                         // 4KB
  __shared__ int   lds_idx[ROWS_PER_WG];
  __shared__ float lds_loss[8];

  const int tid  = threadIdx.x;
  const int wv   = tid >> 5;        // wave 0..7
  const int lane = tid & 31;
  const int m    = lane & 15;       // row-within-tile for A / column n for B/D
  const int hi   = lane >> 4;

  const int rowBlock = blockIdx.x * ROWS_PER_WG;
  const int row = rowBlock + wv * 16 + m;        // flat vector index n
  const int b   = row >> 12;                     // n = b*T + t
  const int t   = row & (T_DIM - 1);

  // async stage of one 64-code chunk (2048 x 16B across 256 threads)
  auto stage = [&](int cc, int buf) {
    const uint4* src = (const uint4*)(cbh + (size_t)cc * CHUNK_HALFS);
#pragma unroll
    for (int i = 0; i < 8; ++i) {
      int s = i * 256 + tid;
#if defined(HAVE_ASYNC_LDS)
      __builtin_amdgcn_global_load_async_to_lds_b128(
          (gvs4i*)(src + s),
          (lvs4i*)&lds_cb[buf][(size_t)s * 8],
          0, 0);
#else
      ((uint4*)lds_cb[buf])[s] = src[s];
#endif
    }
  };
  auto wait_async = [&]() {
#if defined(HAVE_ASYNC_LDS)
#if __has_builtin(__builtin_amdgcn_s_wait_asynccnt)
    __builtin_amdgcn_s_wait_asynccnt(0);
#else
    asm volatile("s_wait_asynccnt 0" ::: "memory");
#endif
#endif
  };

  // kick off chunk 0 before doing unrelated work (A loads hide its latency)
  stage(0, 0);

  // stage |e|^2 table into LDS (used every code tile)
  for (int i = tid; i < K_CODES; i += 256) lds_q[i] = qn[i];

  // ---- load resident A fragments (16x256 f16 per wave) ----
  // A (16x32, 16-bit): lane holds row M=lane&15; element e -> K = 16*hi + e
  const float* xrow = x + ((size_t)b * C_DIM) * T_DIM + t;
  v16h A[8];
#pragma unroll
  for (int kc = 0; kc < 8; ++kc) {
#pragma unroll
    for (int e = 0; e < 16; ++e) {
      int c = kc * 32 + 16 * hi + e;
      A[kc][e] = (_Float16)xrow[(size_t)c * T_DIM];
    }
  }

  float minv[8];
  int   mini[8];
#pragma unroll
  for (int v = 0; v < 8; ++v) { minv[v] = 3.4e38f; mini[v] = 0; }

  wait_async();
  __syncthreads();   // chunk 0 + lds_q visible to all waves

  // ---- sweep codebook: prefetch chunk cc+1 while computing chunk cc ----
#pragma unroll 1
  for (int cc = 0; cc < N_CHUNKS; ++cc) {
    const int buf = cc & 1;
    if (cc + 1 < N_CHUNKS) stage(cc + 1, buf ^ 1);   // overlaps with WMMAs below

    for (int ct4 = 0; ct4 < 4; ++ct4) {              // 4 tiles of 16 codes
      const int ct = cc * 4 + ct4;
      v8f acc = {};
#pragma unroll
      for (int kc = 0; kc < 8; ++kc) {
        v16h Bv = *(const v16h*)&lds_cb[buf][(((ct4 * 8 + kc) * 32) + lane) * 16];
        acc = __builtin_amdgcn_wmma_f32_16x16x32_f16(
            false, A[kc], false, Bv, (short)0, acc, false, false);
      }
      // D layout: VGPR v holds row M = v + 8*hi, column N = lane&15
      const int code = ct * 16 + m;
      const float q = lds_q[code];
#pragma unroll
      for (int v = 0; v < 8; ++v) {
        float d = q - 2.0f * acc[v];
        if (d < minv[v]) { minv[v] = d; mini[v] = code; }
      }
    }

    wait_async();     // prefetch finished (ran concurrently with compute)
    __syncthreads();  // all waves done reading buf; next buffer consistent
  }

  // ---- reduce argmin across the 16 lanes sharing each row ----
#pragma unroll
  for (int off = 8; off >= 1; off >>= 1) {
#pragma unroll
    for (int v = 0; v < 8; ++v) {
      float ov = __shfl_xor(minv[v], off, 32);
      int   oi = __shfl_xor(mini[v], off, 32);
      if (ov < minv[v] || (ov == minv[v] && oi < mini[v])) {
        minv[v] = ov; mini[v] = oi;
      }
    }
  }
  if (m == 0) {
#pragma unroll
    for (int v = 0; v < 8; ++v) lds_idx[wv * 16 + 8 * hi + v] = mini[v];
  }
  __syncthreads();

  // ---- gather + write quant [B,C,T], fused loss accumulation ----
  const int tl = tid & 127;            // local row (consecutive t)
  const int ch = tid >> 7;             // c parity handled by thread half
  const int wrow = rowBlock + tl;
  const int wb = wrow >> 12;
  const int wt = wrow & (T_DIM - 1);
  const int idx = lds_idx[tl];
  const float* crow = cb + (size_t)idx * C_DIM;
  const size_t base = ((size_t)wb * C_DIM) * T_DIM + wt;
  float lacc = 0.0f;
#pragma unroll 4
  for (int i = 0; i < 128; ++i) {
    int c = 2 * i + ch;
    float qv = crow[c];
    size_t o = base + (size_t)c * T_DIM;
    out[o] = qv;
    float dd = qv - x[o];
    lacc += dd * dd;
  }

  // deterministic block loss reduction -> workspace partial
#pragma unroll
  for (int off = 16; off; off >>= 1) lacc += __shfl_xor(lacc, off, 32);
  if (lane == 0) lds_loss[wv] = lacc;
  __syncthreads();
  if (tid == 0) {
    float s = 0.0f;
#pragma unroll
    for (int i = 0; i < 8; ++i) s += lds_loss[i];
    part[blockIdx.x] = s;
  }
}

// ------------------------- final loss sum -----------------------------------
__global__ __launch_bounds__(512) void vq_loss_final(
    const float* __restrict__ part, float* __restrict__ loss) {
  __shared__ float s[512];
  s[threadIdx.x] = part[threadIdx.x];
  __syncthreads();
  for (int st = 256; st > 0; st >>= 1) {
    if ((int)threadIdx.x < st) s[threadIdx.x] += s[threadIdx.x + st];
    __syncthreads();
  }
  if (threadIdx.x == 0)
    *loss = s[0] * (2.0f / (float)(B_DIM * C_DIM * T_DIM));
}

// ---------------------------------------------------------------------------
extern "C" void kernel_launch(void* const* d_in, const int* in_sizes, int n_in,
                              void* d_out, int out_size, void* d_ws, size_t ws_size,
                              hipStream_t stream) {
  const float* x  = (const float*)d_in[0];   // [16,256,4096]
  const float* cb = (const float*)d_in[1];   // [1024,256]
  float* out  = (float*)d_out;               // 16777216 quant + 1 loss
  float* loss = out + (size_t)B_DIM * C_DIM * T_DIM;

  _Float16* cbh = (_Float16*)d_ws;                                   // 512 KB
  float* qn   = (float*)((char*)d_ws + 512 * 1024);                  // 4 KB
  float* part = (float*)((char*)d_ws + 512 * 1024 + 4 * 1024);       // 2 KB

  vq_prep_scramble<<<K_CODES * C_DIM / 256, 256, 0, stream>>>(cb, cbh);
  vq_prep_qnorm<<<K_CODES, 64, 0, stream>>>(cb, qn);
  vq_main<<<N_ROWS / ROWS_PER_WG, 256, 0, stream>>>(x, cb, cbh, qn, out, part);
  vq_loss_final<<<1, 512, 0, stream>>>(part, loss);
}
